// RelativeMultiHeadAttention_29231547417062
// MI455X (gfx1250) — matrix operand order repaired and checked by reading, not compiled, not run on hardware
//
#include <hip/hip_runtime.h>

typedef __bf16 bf16;
typedef __attribute__((ext_vector_type(16))) __bf16 bf16x16;
typedef __attribute__((ext_vector_type(8)))  __bf16 bf16x8;
typedef __attribute__((ext_vector_type(4)))  __bf16 bf16x4;
typedef __attribute__((ext_vector_type(8)))  float  f32x8;

#define LDX 520    // x / ctx row stride (bf16 elems), 512 + 8 pad
#define LDH 72     // per-head 64 + 8 pad
#define NW  262144 // 512*512

static __device__ __forceinline__ bf16x16 frag2(const bf16* p0, const bf16* p1) {
  union { bf16x16 v; bf16x8 h[2]; } u;
  u.h[0] = *(const bf16x8*)p0;
  u.h[1] = *(const bf16x8*)p1;
  return u.v;
}

static __device__ __forceinline__ f32x8 wmma_bf(bf16x16 a, bf16x16 b, f32x8 c) {
  // D = A*B + C, 16x16x32 bf16 -> f32
  return __builtin_amdgcn_wmma_f32_16x16x32_bf16(false, a, false, b, (short)0, c,
                                                 false, false);
}

static __device__ __forceinline__ bf16x4 cvt4(float4 f) {
  bf16x4 r;
  r.x = (bf16)f.x; r.y = (bf16)f.y; r.z = (bf16)f.z; r.w = (bf16)f.w;
  return r;
}

__global__ void convert_weights_kernel(const float4* __restrict__ wq,
                                       const float4* __restrict__ wk,
                                       const float4* __restrict__ wv,
                                       const float4* __restrict__ wo,
                                       bf16x4* __restrict__ dst) {
  int i = blockIdx.x * 256 + threadIdx.x;
  const int N4 = NW / 4;
  if (i < N4) {
    dst[i]          = cvt4(wq[i]);
    dst[i + N4]     = cvt4(wk[i]);
    dst[i + 2 * N4] = cvt4(wv[i]);
    dst[i + 3 * N4] = cvt4(wo[i]);
  }
}

__global__ __launch_bounds__(256) void relmha_kernel(
    const float* __restrict__ x,       // (B,64,512)
    const float* __restrict__ bq,
    const float* __restrict__ bk,
    const float* __restrict__ bv,
    const float* __restrict__ bo,
    const float* __restrict__ rel_tab, // (225,8)
    const bf16*  __restrict__ wq,      // (512,512) bf16, row n, col k
    const bf16*  __restrict__ wk,
    const bf16*  __restrict__ wv,
    const bf16*  __restrict__ wo,
    float* __restrict__ out)           // (B,64,512)
{
  extern __shared__ char smem_raw[];
  bf16*  xs  = (bf16*)smem_raw;        // 64 x LDX  (66560 B)
  bf16*  ctx = xs  + 64 * LDX;         // 64 x LDX  (66560 B)
  bf16*  qs  = ctx + 64 * LDX;         // 64 x LDH  (9216 B)
  bf16*  ks  = qs  + 64 * LDH;
  bf16*  vt  = ks  + 64 * LDH;         // v stored TRANSPOSED: vt[d][j]
  bf16*  ps  = vt  + 64 * LDH;         // softmax probs
  float* ss  = (float*)(ps + 64 * LDH);// 64 x LDH f32 scores (18432 B)
  float* rel = ss + 64 * LDH;          // 225*8 f32 (7200 B)

  const int b    = blockIdx.x;
  const int tid  = threadIdx.x;
  const int wave = tid >> 5;
  const int lane = tid & 31;
  const int half = lane >> 4;
  const int lr   = lane & 15;

  // ---- stage x_b into LDS as bf16 (vectorized: float4 -> bf16x4) ----
  const float4* xb4 = (const float4*)(x + (size_t)b * (64 * 512));
  for (int i = tid; i < 64 * 128; i += 256) {
    int m = i >> 7, j = i & 127;
    *(bf16x4*)(xs + m * LDX + j * 4) = cvt4(xb4[i]);
  }
  for (int i = tid; i < 225 * 8; i += 256) rel[i] = rel_tab[i];
  __syncthreads();

  // 64x64-stage tile mapping: wave owns ONE n-tile and TWO m-tiles so that the
  // (L2-sourced) B fragment is loaded once and reused for 2 WMMAs.
  const int n0  = (wave & 3) * 16;          // n-tile
  const int m0a = ((wave >> 2) * 2) * 16;   // first m-tile
  const int m0b = m0a + 16;                 // second m-tile
  const int mrowA = m0a + half * 8;
  const int mrowB = m0b + half * 8;

  const float scale = 0.125f; // 1/sqrt(64)

  for (int h = 0; h < 8; ++h) {
    // ================= Q / K / V projections (M=64,N=64,K=512) =============
    const bf16*  Ws[3] = {wq, wk, wv};
    const float* Bs[3] = {bq, bk, bv};
#pragma unroll
    for (int mat = 0; mat < 3; ++mat) {
      const bf16* Wb = Ws[mat] + (size_t)(h * 64) * 512;
      f32x8 acc0 = {}, acc1 = {};
      for (int k0 = 0; k0 < 512; k0 += 32) {
        const bf16* br = Wb + (size_t)(n0 + lr) * 512 + k0 + half * 8;
        bf16x16 bf = frag2(br, br + 16);
        const bf16* ar0 = xs + (m0a + lr) * LDX + k0 + half * 8;
        const bf16* ar1 = xs + (m0b + lr) * LDX + k0 + half * 8;
        acc0 = wmma_bf(frag2(ar0, ar0 + 16), bf, acc0);
        acc1 = wmma_bf(frag2(ar1, ar1 + 16), bf, acc1);
      }
      const float bb = Bs[mat][h * 64 + n0 + lr];
      if (mat == 2) {
        // store V transposed: vt[d][j]
#pragma unroll
        for (int e = 0; e < 8; ++e) {
          vt[(n0 + lr) * LDH + mrowA + e] = (bf16)(acc0[e] + bb);
          vt[(n0 + lr) * LDH + mrowB + e] = (bf16)(acc1[e] + bb);
        }
      } else {
        bf16* dst = (mat == 0) ? qs : ks;
#pragma unroll
        for (int e = 0; e < 8; ++e) {
          dst[(mrowA + e) * LDH + n0 + lr] = (bf16)(acc0[e] + bb);
          dst[(mrowB + e) * LDH + n0 + lr] = (bf16)(acc1[e] + bb);
        }
      }
    }
    __syncthreads();

    // ================= scores = q @ k^T  (M=N=K=64) =========================
    {
      f32x8 acc0 = {}, acc1 = {};
#pragma unroll
      for (int k0 = 0; k0 < 64; k0 += 32) {
        const bf16* br = ks + (n0 + lr) * LDH + k0 + half * 8;  // row j of k
        bf16x16 bf = frag2(br, br + 16);
        const bf16* ar0 = qs + (m0a + lr) * LDH + k0 + half * 8;
        const bf16* ar1 = qs + (m0b + lr) * LDH + k0 + half * 8;
        acc0 = wmma_bf(frag2(ar0, ar0 + 16), bf, acc0);
        acc1 = wmma_bf(frag2(ar1, ar1 + 16), bf, acc1);
      }
#pragma unroll
      for (int e = 0; e < 8; ++e) {
        ss[(mrowA + e) * LDH + n0 + lr] = acc0[e];
        ss[(mrowB + e) * LDH + n0 + lr] = acc1[e];
      }
    }
    __syncthreads();

    // ================= softmax + relative bias (4 lanes per row) ============
    {
      const int i  = tid >> 2;          // row 0..63
      const int j0 = (tid & 3) * 16;    // 16-col slice
      const int ri = i >> 3, fi = i & 7;
      float pv[16];
      float mx = -1e30f;
#pragma unroll
      for (int jj = 0; jj < 16; ++jj) {
        int j = j0 + jj;
        int rj = j >> 3, fj = j & 7;
        int idx = (ri - rj + 7) * 15 + (fi - fj + 7);
        float s = ss[i * LDH + j] * scale + rel[idx * 8 + h];
        pv[jj] = s;
        mx = fmaxf(mx, s);
      }
      mx = fmaxf(mx, __shfl_xor(mx, 1));
      mx = fmaxf(mx, __shfl_xor(mx, 2));
      float sum = 0.f;
#pragma unroll
      for (int jj = 0; jj < 16; ++jj) {
        pv[jj] = __expf(pv[jj] - mx);
        sum += pv[jj];
      }
      sum += __shfl_xor(sum, 1);
      sum += __shfl_xor(sum, 2);
      float inv = 1.0f / sum;
#pragma unroll
      for (int jj = 0; jj < 16; ++jj)
        ps[i * LDH + j0 + jj] = (bf16)(pv[jj] * inv);
    }
    __syncthreads();

    // ================= ctx[:, h*64:] = P @ V  (M=N=K=64) ====================
    {
      f32x8 acc0 = {}, acc1 = {};
#pragma unroll
      for (int k0 = 0; k0 < 64; k0 += 32) {
        const bf16* br = vt + (n0 + lr) * LDH + k0 + half * 8;  // vt[d][j]
        bf16x16 bf = frag2(br, br + 16);
        const bf16* ar0 = ps + (m0a + lr) * LDH + k0 + half * 8;
        const bf16* ar1 = ps + (m0b + lr) * LDH + k0 + half * 8;
        acc0 = wmma_bf(frag2(ar0, ar0 + 16), bf, acc0);
        acc1 = wmma_bf(frag2(ar1, ar1 + 16), bf, acc1);
      }
#pragma unroll
      for (int e = 0; e < 8; ++e) {
        ctx[(mrowA + e) * LDX + h * 64 + n0 + lr] = (bf16)acc0[e];
        ctx[(mrowB + e) * LDX + h * 64 + n0 + lr] = (bf16)acc1[e];
      }
    }
    __syncthreads();
  } // heads

  // ================= out = ctx @ Wo^T + bo  (M=64,N=512,K=512) ==============
  // Wave owns 4 n-tiles (nt = 4w..4w+3) and ALL 4 m-tiles per n-tile, so each
  // Wo fragment is fetched from L2 exactly once per workgroup (4 WMMAs per B).
  {
    float* ob = out + (size_t)b * (64 * 512);
#pragma unroll
    for (int j = 0; j < 4; ++j) {
      const int fn0 = (wave * 4 + j) * 16;
      f32x8 acc[4] = {{}, {}, {}, {}};
      for (int k0 = 0; k0 < 512; k0 += 32) {
        const bf16* br = wo + (size_t)(fn0 + lr) * 512 + k0 + half * 8;
        bf16x16 bf = frag2(br, br + 16);
#pragma unroll
        for (int mt = 0; mt < 4; ++mt) {
          const bf16* ar = ctx + (mt * 16 + lr) * LDX + k0 + half * 8;
          acc[mt] = wmma_bf(frag2(ar, ar + 16), bf, acc[mt]);
        }
      }
      const float bb = bo[fn0 + lr];
#pragma unroll
      for (int mt = 0; mt < 4; ++mt) {
        const int fmrow = mt * 16 + half * 8;
#pragma unroll
        for (int e = 0; e < 8; ++e)
          ob[(fmrow + e) * 512 + fn0 + lr] = acc[mt][e] + bb;
      }
    }
  }
}

extern "C" void kernel_launch(void* const* d_in, const int* in_sizes, int n_in,
                              void* d_out, int out_size, void* d_ws, size_t ws_size,
                              hipStream_t stream) {
  const float* x   = (const float*)d_in[0];
  const float* Wq  = (const float*)d_in[1];
  const float* bq  = (const float*)d_in[2];
  const float* Wk  = (const float*)d_in[3];
  const float* bk  = (const float*)d_in[4];
  const float* Wv  = (const float*)d_in[5];
  const float* bv  = (const float*)d_in[6];
  const float* Wo  = (const float*)d_in[7];
  const float* bo  = (const float*)d_in[8];
  const float* rel = (const float*)d_in[9];
  float* out = (float*)d_out;

  bf16* wbf = (bf16*)d_ws;  // 4 * 512*512 bf16 = 2 MB

  convert_weights_kernel<<<(NW / 4 + 255) / 256, 256, 0, stream>>>(
      (const float4*)Wq, (const float4*)Wk, (const float4*)Wv, (const float4*)Wo,
      (bf16x4*)wbf);

  const int B = in_sizes[0] / (64 * 512);
  const size_t shmem =
      (size_t)(2 * 64 * LDX) * sizeof(bf16) +   // xs + ctx
      (size_t)(4 * 64 * LDH) * sizeof(bf16) +   // qs, ks, vt, ps
      (size_t)(64 * LDH) * sizeof(float) +      // ss
      (size_t)(225 * 8) * sizeof(float);        // rel

  (void)hipFuncSetAttribute((const void*)relmha_kernel,
                            hipFuncAttributeMaxDynamicSharedMemorySize,
                            (int)shmem);

  relmha_kernel<<<B, 256, shmem, stream>>>(x, bq, bk, bv, bo, rel,
                                           wbf, wbf + NW, wbf + 2 * NW, wbf + 3 * NW,
                                           out);
}
